// MOF_Net_27230092657067
// MI455X (gfx1250) — compile-verified, matrix operand order checked by reading, not run on hardware
//
#include <hip/hip_runtime.h>

typedef __bf16 bf16_t;
typedef __attribute__((ext_vector_type(16))) __bf16 v16bf;
typedef __attribute__((ext_vector_type(8)))  float  v8f;

__device__ __forceinline__ v8f wmma_bf16(v16bf a, v16bf b, v8f c) {
  return __builtin_amdgcn_wmma_f32_16x16x32_bf16(false, a, false, b, (short)0, c,
                                                 false, false);
}

// Non-returning device-scope f32 atomic add (guaranteed native instruction).
__device__ __forceinline__ void global_fadd(float* p, float v) {
  asm volatile("global_atomic_add_f32 %0, %1, off scope:SCOPE_DEV"
               :: "v"(p), "v"(v) : "memory");
}

// Single-instruction ReLU (avoid fmaxf's canonicalize+max pair).
__device__ __forceinline__ float relu(float x) { return x > 0.0f ? x : 0.0f; }

// bf16 A/B fragment slot -> K offset within a 32-wide K chunk.
// Lanes 0-15 hold K 0..7 (slots 0-7) and 16..23 (slots 8-15);
// lanes 16-31 hold K 8..15 and 24..31.   (ISA 7.12.2, 16-bit A 16x32)
__device__ __forceinline__ int klocal(int i, int half) {
  return (i < 8) ? (8 * half + i) : (16 + 8 * half + (i - 8));
}

__global__ void zero_f32(float* __restrict__ p, long long n) {
  long long i = (long long)blockIdx.x * blockDim.x + threadIdx.x;
  long long s = (long long)gridDim.x * blockDim.x;
  for (; i < n; i += s) p[i] = 0.0f;
}

// ---------------- Stage 1: DGCN edge MLP (48->64->relu->32) + scatter ----------
__global__ __launch_bounds__(256) void dgcn_edge_kernel(
    const float* __restrict__ x2, const int* __restrict__ ei2,
    const float* __restrict__ ea2,
    const float* __restrict__ W2a, const float* __restrict__ b2a,
    const float* __restrict__ W2b, const float* __restrict__ b2b,
    float* __restrict__ e_t, int E2)
{
  __shared__ float Hs[8][16 * 68];          // per-wave 16x64 hidden tile (padded)
  const int lane = threadIdx.x & 31;
  const int wv   = threadIdx.x >> 5;
  const int r    = lane & 15;
  const int half = lane >> 4;
  const int gw   = (int)((blockIdx.x * blockDim.x + threadIdx.x) >> 5);
  const int nw   = (int)((gridDim.x * blockDim.x) >> 5);
  const int ntiles = (E2 + 15) >> 4;
  const int* s2 = ei2;
  const int* d2 = ei2 + E2;

  // Weight fragments, built once, kept in VGPRs.
  v16bf wA[2][4];                           // W2a: K=48 (pad to 64), N=64
  #pragma unroll
  for (int kc = 0; kc < 2; ++kc)
    #pragma unroll
    for (int nt = 0; nt < 4; ++nt) {
      const int n = nt * 16 + r;
      #pragma unroll
      for (int i = 0; i < 16; ++i) {
        const int k = kc * 32 + klocal(i, half);
        wA[kc][nt][i] = (k < 48) ? (bf16_t)W2a[k * 64 + n] : (bf16_t)0.0f;
      }
    }
  v16bf wB[2][2];                           // W2b: K=64, N=32
  #pragma unroll
  for (int kc = 0; kc < 2; ++kc)
    #pragma unroll
    for (int nt = 0; nt < 2; ++nt) {
      const int n = nt * 16 + r;
      #pragma unroll
      for (int i = 0; i < 16; ++i) {
        const int k = kc * 32 + klocal(i, half);
        wB[kc][nt][i] = (bf16_t)W2b[k * 32 + n];
      }
    }
  float biasA[4], biasB[2];
  #pragma unroll
  for (int nt = 0; nt < 4; ++nt) biasA[nt] = b2a[nt * 16 + r];
  #pragma unroll
  for (int nt = 0; nt < 2; ++nt) biasB[nt] = b2b[nt * 16 + r];

  float* H = Hs[wv];

  for (int t = gw; t < ntiles; t += nw) {
    const int base = t << 4;
    int e_cl = base + r; if (e_cl > E2 - 1) e_cl = E2 - 1;   // clamped A-row edge
    const int src = s2[e_cl];
    const float* xr = x2  + (size_t)src  * 32;
    const float* er = ea2 + (size_t)e_cl * 16;

    v16bf a0, a1;                            // K chunk 0: x2 row; chunk 1: attr+pad
    #pragma unroll
    for (int i = 0; i < 16; ++i) {
      const int k = klocal(i, half);
      a0[i] = (bf16_t)xr[k];
      a1[i] = (k < 16) ? (bf16_t)er[k] : (bf16_t)0.0f;
    }

    // Layer 1 + ReLU -> LDS hidden tile
    #pragma unroll
    for (int nt = 0; nt < 4; ++nt) {
      v8f c;
      #pragma unroll
      for (int j = 0; j < 8; ++j) c[j] = biasA[nt];
      c = wmma_bf16(a0, wA[0][nt], c);
      c = wmma_bf16(a1, wA[1][nt], c);
      const int n = nt * 16 + r;
      #pragma unroll
      for (int j = 0; j < 8; ++j)
        H[(half * 8 + j) * 68 + n] = relu(c[j]);
    }
    asm volatile("s_wait_dscnt 0" ::: "memory");

    v16bf h0, h1;                            // hidden as A fragments (K=64)
    #pragma unroll
    for (int i = 0; i < 16; ++i) {
      const int k = klocal(i, half);
      h0[i] = (bf16_t)H[r * 68 + k];
      h1[i] = (bf16_t)H[r * 68 + 32 + k];
    }

    v8f c2[2];
    #pragma unroll
    for (int nt = 0; nt < 2; ++nt) {
      v8f c;
      #pragma unroll
      for (int j = 0; j < 8; ++j) c[j] = biasB[nt];
      c = wmma_bf16(h0, wB[0][nt], c);
      c = wmma_bf16(h1, wB[1][nt], c);
      c2[nt] = c;
    }

    // Scatter-add msg2 rows to e_t[dst] (after all WMMAs -> EXEC full during WMMA)
    #pragma unroll
    for (int j = 0; j < 8; ++j) {
      const int e = base + half * 8 + j;
      if (e < E2) {
        const int dst = d2[e];
        float* o = e_t + (size_t)dst * 32 + r;
        global_fadd(o,      c2[0][j]);
        global_fadd(o + 16, c2[1][j]);
      }
    }
  }
}

// ------- Stage 2: MOLGCN edge MLP (64->64->relu->32) + fused graph pooling ------
__global__ __launch_bounds__(256) void molgcn_pool_kernel(
    const float* __restrict__ x1, const int* __restrict__ ei1,
    const int* __restrict__ xbatch, const float* __restrict__ e_t,
    const float* __restrict__ W1a, const float* __restrict__ b1a,
    const float* __restrict__ W1b, const float* __restrict__ b1b,
    float* __restrict__ out, int E1)
{
  __shared__ float Hs[8][16 * 68];
  __shared__ float pool[64 * 32];            // per-block pooled accumulator
  for (int i = threadIdx.x; i < 64 * 32; i += 256) pool[i] = 0.0f;
  __syncthreads();

  const int lane = threadIdx.x & 31;
  const int wv   = threadIdx.x >> 5;
  const int r    = lane & 15;
  const int half = lane >> 4;
  const int gw   = (int)((blockIdx.x * blockDim.x + threadIdx.x) >> 5);
  const int nw   = (int)((gridDim.x * blockDim.x) >> 5);
  const int ntiles = (E1 + 15) >> 4;
  const int* s1 = ei1;
  const int* d1 = ei1 + E1;

  v16bf wA[2][4];                            // W1a: K=64, N=64
  #pragma unroll
  for (int kc = 0; kc < 2; ++kc)
    #pragma unroll
    for (int nt = 0; nt < 4; ++nt) {
      const int n = nt * 16 + r;
      #pragma unroll
      for (int i = 0; i < 16; ++i) {
        const int k = kc * 32 + klocal(i, half);
        wA[kc][nt][i] = (bf16_t)W1a[k * 64 + n];
      }
    }
  v16bf wB[2][2];                            // W1b: K=64, N=32
  #pragma unroll
  for (int kc = 0; kc < 2; ++kc)
    #pragma unroll
    for (int nt = 0; nt < 2; ++nt) {
      const int n = nt * 16 + r;
      #pragma unroll
      for (int i = 0; i < 16; ++i) {
        const int k = kc * 32 + klocal(i, half);
        wB[kc][nt][i] = (bf16_t)W1b[k * 32 + n];
      }
    }
  float biasA[4], biasB[2];
  #pragma unroll
  for (int nt = 0; nt < 4; ++nt) biasA[nt] = b1a[nt * 16 + r];
  #pragma unroll
  for (int nt = 0; nt < 2; ++nt) biasB[nt] = b1b[nt * 16 + r];

  float* H = Hs[wv];

  for (int t = gw; t < ntiles; t += nw) {
    const int base = t << 4;
    int e_cl = base + r; if (e_cl > E1 - 1) e_cl = E1 - 1;
    const int src = s1[e_cl];
    const float* xr = x1  + (size_t)src  * 32;
    const float* tr = e_t + (size_t)e_cl * 32;     // line graph: e_t row == edge id

    v16bf a0, a1;
    #pragma unroll
    for (int i = 0; i < 16; ++i) {
      const int k = klocal(i, half);
      a0[i] = (bf16_t)xr[k];
      a1[i] = (bf16_t)tr[k];
    }

    #pragma unroll
    for (int nt = 0; nt < 4; ++nt) {
      v8f c;
      #pragma unroll
      for (int j = 0; j < 8; ++j) c[j] = biasA[nt];
      c = wmma_bf16(a0, wA[0][nt], c);
      c = wmma_bf16(a1, wA[1][nt], c);
      const int n = nt * 16 + r;
      #pragma unroll
      for (int j = 0; j < 8; ++j)
        H[(half * 8 + j) * 68 + n] = relu(c[j]);
    }
    asm volatile("s_wait_dscnt 0" ::: "memory");

    v16bf h0, h1;
    #pragma unroll
    for (int i = 0; i < 16; ++i) {
      const int k = klocal(i, half);
      h0[i] = (bf16_t)H[r * 68 + k];
      h1[i] = (bf16_t)H[r * 68 + 32 + k];
    }

    v8f c2[2];
    #pragma unroll
    for (int nt = 0; nt < 2; ++nt) {
      v8f c;
      #pragma unroll
      for (int j = 0; j < 8; ++j) c[j] = biasB[nt];
      c = wmma_bf16(h0, wB[0][nt], c);
      c = wmma_bf16(h1, wB[1][nt], c);
      c2[nt] = c;
    }

    // pooled[g] = sum over edges of msg1 at nodes of graph g -> accumulate in LDS
    #pragma unroll
    for (int j = 0; j < 8; ++j) {
      const int e = base + half * 8 + j;
      if (e < E1) {
        const int g = xbatch[d1[e]];
        atomicAdd(&pool[g * 32 + r],      c2[0][j]);
        atomicAdd(&pool[g * 32 + 16 + r], c2[1][j]);
      }
    }
  }

  __syncthreads();
  for (int i = threadIdx.x; i < 64 * 32; i += 256)
    global_fadd(out + i, pool[i] * 0.5f);          // includes the /2
}

extern "C" void kernel_launch(void* const* d_in, const int* in_sizes, int n_in,
                              void* d_out, int out_size, void* d_ws, size_t ws_size,
                              hipStream_t stream) {
  const float* x1     = (const float*)d_in[0];
  const float* x2     = (const float*)d_in[1];
  const int*   ei1    = (const int*)d_in[2];
  const int*   ei2    = (const int*)d_in[3];
  const int*   xbatch = (const int*)d_in[4];
  const float* ea2    = (const float*)d_in[5];
  const float* W2a = (const float*)d_in[6];
  const float* b2a = (const float*)d_in[7];
  const float* W2b = (const float*)d_in[8];
  const float* b2b = (const float*)d_in[9];
  const float* W1a = (const float*)d_in[10];
  const float* b1a = (const float*)d_in[11];
  const float* W1b = (const float*)d_in[12];
  const float* b1b = (const float*)d_in[13];

  const int N2 = in_sizes[1] / 32;
  const int E1 = in_sizes[2] / 2;
  const int E2 = in_sizes[3] / 2;

  float* e_t = (float*)d_ws;                // [N2, 32] scratch
  float* out = (float*)d_out;               // [64, 32]
  (void)ws_size; (void)n_in;

  zero_f32<<<1024, 256, 0, stream>>>(e_t, (long long)N2 * 32);
  zero_f32<<<1, 256, 0, stream>>>(out, (long long)out_size);

  int tiles2  = (E2 + 15) / 16;
  int blocks1 = (tiles2 + 7) / 8; if (blocks1 > 2048) blocks1 = 2048;
  dgcn_edge_kernel<<<blocks1, 256, 0, stream>>>(x2, ei2, ea2, W2a, b2a, W2b, b2b,
                                                e_t, E2);

  int tiles1  = (E1 + 15) / 16;
  int blocks2 = (tiles1 + 7) / 8; if (blocks2 > 2048) blocks2 = 2048;
  molgcn_pool_kernel<<<blocks2, 256, 0, stream>>>(x1, ei1, xbatch, e_t,
                                                  W1a, b1a, W1b, b1b, out, E1);
}
